// Head_52905407152260
// MI455X (gfx1250) — matrix operand order repaired and checked by reading, not compiled, not run on hardware
//
#include <hip/hip_runtime.h>
#include <hip/hip_bf16.h>

typedef __attribute__((ext_vector_type(16))) __bf16 v16bf;
typedef __attribute__((ext_vector_type(8)))  __bf16 v8bf;
typedef __attribute__((ext_vector_type(4)))  __bf16 v4bf;
typedef __attribute__((ext_vector_type(8)))  float  v8f;
typedef int v4i_t __attribute__((ext_vector_type(4)));

#define N_EMBD 256
#define HEAD   64
#define TSEQ   4096
#define BATCH  4

// Async global->LDS copy path (gfx1250 GLOBAL_LOAD_ASYNC_TO_LDS_B128).
#if defined(__gfx1250__) && __has_builtin(__builtin_amdgcn_global_load_async_to_lds_b128)
#define HAVE_ASYNC_LDS 1
#else
#define HAVE_ASYNC_LDS 0
#endif

static __device__ __forceinline__ v8f wmma_bf16(v16bf a, v16bf b, v8f c) {
  return __builtin_amdgcn_wmma_f32_16x16x32_bf16(
      false, a, false, b, (short)0, c, false, false);
}

union frag16 { v16bf v; v8bf h[2]; };

// Build a v16bf fragment from two contiguous 8-element (16B) LDS runs.
static __device__ __forceinline__ v16bf ld_frag(const __bf16* p0, const __bf16* p1) {
  frag16 f;
  f.h[0] = *(const v8bf*)p0;   // ds_load_b128
  f.h[1] = *(const v8bf*)p1;   // ds_load_b128
  return f.v;
}

// ---------------------------------------------------------------------------
// Kernel 1: q = xWq + bq, k = xWk + bk, v = xWv  (bf16 outputs to workspace)
// ---------------------------------------------------------------------------
__global__ __launch_bounds__(256) void qkv_proj_kernel(
    const float* __restrict__ x,
    const float* __restrict__ Wq, const float* __restrict__ bq,
    const float* __restrict__ Wk, const float* __restrict__ bk,
    const float* __restrict__ Wv,
    __bf16* __restrict__ qout, __bf16* __restrict__ kout,
    __bf16* __restrict__ vout)
{
  // strides multiple of 8 elems, rows 16B-aligned
  __shared__ alignas(16) __bf16 xs[64][40];     // 64 rows x 32 K
  __shared__ alignas(16) __bf16 wshT[192][40];  // 192 cols x 32 K (transposed)

  const int tid   = threadIdx.x;
  const int lane  = tid & 31;
  const int wave  = tid >> 5;
  const int n     = lane & 15;
  const int khalf = lane >> 4;
  const int k0h   = khalf * 8;
  const size_t row0 = (size_t)blockIdx.x * 64;

  v8f acc[6] = {};

  for (int k0 = 0; k0 < N_EMBD; k0 += 32) {
    __syncthreads();
    // stage x chunk: float4 global loads -> packed v4bf LDS stores
    for (int e = tid * 4; e < 64 * 32; e += 256 * 4) {
      int r = e >> 5, c = e & 31;
      float4 xv = *(const float4*)(x + (row0 + r) * N_EMBD + k0 + c);
      v4bf o;
      o[0] = (__bf16)xv.x; o[1] = (__bf16)xv.y;
      o[2] = (__bf16)xv.z; o[3] = (__bf16)xv.w;
      *(v4bf*)&xs[r][c] = o;  // ds_store_b64
    }
    // stage W chunk transposed: float4 loads, scalar transposed stores
    for (int e = tid * 4; e < 32 * 192; e += 256 * 4) {
      int r = e / 192, c = e % 192;           // c multiple of 4, same row
      int p = c >> 6, cc = c & 63;
      const float* wsrc = (p == 0) ? Wq : ((p == 1) ? Wk : Wv);
      float4 wv = *(const float4*)(wsrc + (k0 + r) * HEAD + cc);
      wshT[c + 0][r] = (__bf16)wv.x;
      wshT[c + 1][r] = (__bf16)wv.y;
      wshT[c + 2][r] = (__bf16)wv.z;
      wshT[c + 3][r] = (__bf16)wv.w;
    }
    __syncthreads();

    #pragma unroll
    for (int i = 0; i < 6; ++i) {
      int tt = wave + 8 * i;
      int ti = tt & 3;
      int tj = tt >> 2;
      v16bf a = ld_frag(&xs[ti * 16 + n][k0h], &xs[ti * 16 + n][k0h + 16]);
      v16bf b = ld_frag(&wshT[tj * 16 + n][k0h],
                        &wshT[tj * 16 + n][k0h + 16]);
      acc[i] = wmma_bf16(a, b, acc[i]);
    }
  }

  // bias + store (bf16)
  #pragma unroll
  for (int i = 0; i < 6; ++i) {
    int tt = wave + 8 * i;
    int ti = tt & 3, tj = tt >> 2;
    int p = tj >> 2;                 // 0=q,1=k,2=v
    int col = (tj & 3) * 16 + n;
    float bias = 0.f;
    if (p == 0) bias = bq[col];
    else if (p == 1) bias = bk[col];
    __bf16* outp = (p == 0) ? qout : ((p == 1) ? kout : vout);
    #pragma unroll
    for (int g = 0; g < 8; ++g) {
      int m = ti * 16 + g + khalf * 8;
      outp[(row0 + m) * HEAD + col] = (__bf16)(acc[i][g] + bias);
    }
  }
}

// ---------------------------------------------------------------------------
// Kernel 2: flash attention (no mask). One block = (batch, 64-row q tile).
// ---------------------------------------------------------------------------
__global__ __launch_bounds__(256) void attn_kernel(
    const __bf16* __restrict__ qws, const __bf16* __restrict__ kws,
    const __bf16* __restrict__ vws, float* __restrict__ out)
{
  __shared__ alignas(16) __bf16 qs[64][72];   // [q row][hk]
  __shared__ alignas(16) __bf16 ks[64][72];   // [key row][hk]
  __shared__ alignas(16) __bf16 vsT[64][72];  // [h][key row]  (transposed!)
  __shared__ alignas(16) __bf16 Ps[64][72];   // [q row][key]
  __shared__ alignas(16) float  Ss[64][68];   // [q row][key]
  __shared__ float mrow[64], lrow[64], crow[64];

  const int tid   = threadIdx.x;
  const int lane  = tid & 31;
  const int wave  = tid >> 5;
  const int n     = lane & 15;
  const int khalf = lane >> 4;
  const int k0h   = khalf * 8;

  const int bb = blockIdx.x / (TSEQ / 64);
  const int q0 = (blockIdx.x % (TSEQ / 64)) * 64;
  const size_t qbase   = ((size_t)bb * TSEQ + q0) * HEAD;
  const size_t kvbase0 = (size_t)bb * TSEQ * HEAD;

  // Q tile, prescaled by C^-0.5 = 1/16 (exact in bf16)
  for (int e = tid * 8; e < 64 * HEAD; e += 256 * 8) {
    int r = e >> 6, c = e & 63;
    v8bf qv = *(const v8bf*)(qws + qbase + r * HEAD + c);  // global_load_b128
    v8bf qo;
    #pragma unroll
    for (int q = 0; q < 8; ++q) qo[q] = (__bf16)((float)qv[q] * 0.0625f);
    *(v8bf*)&qs[r][c] = qo;                                // ds_store_b128
  }
  if (tid < 64) { mrow[tid] = -3.0e38f; lrow[tid] = 0.f; }

  v8f accO[2] = {};
  const int ti0 = wave & 3,       tj0 = wave >> 2;
  const int ti1 = (wave + 8) & 3, tj1 = (wave + 8) >> 2;

  for (int j0 = 0; j0 < TSEQ; j0 += 64) {
    __syncthreads();  // previous iteration done with ks/vsT/Ps (and init)
    const size_t kvb = kvbase0 + (size_t)j0 * HEAD;

#if HAVE_ASYNC_LDS
    // K tile: pure copy -> async DMA into LDS, overlapped with V transpose.
    #pragma unroll
    for (int e = tid; e < 512; e += 256) {      // 512 x 16B chunks = 8 KB
      int r = e >> 3, c = (e & 7) * 8;
      __builtin_amdgcn_global_load_async_to_lds_b128(
          (__attribute__((address_space(1))) v4i_t*)(kws + kvb + r * HEAD + c),
          (__attribute__((address_space(3))) v4i_t*)&ks[r][c],
          0, 0);
    }
    for (int e = tid * 8; e < 64 * HEAD; e += 256 * 8) {
      int r = e >> 6, c = e & 63;
      v8bf vv = *(const v8bf*)(vws + kvb + r * HEAD + c);
      #pragma unroll
      for (int q = 0; q < 8; ++q) vsT[c + q][r] = vv[q];   // transpose store
    }
#if __has_builtin(__builtin_amdgcn_s_wait_asynccnt)
    __builtin_amdgcn_s_wait_asynccnt(0);
#else
    asm volatile("s_wait_asynccnt 0" ::: "memory");
#endif
#else
    for (int e = tid * 8; e < 64 * HEAD; e += 256 * 8) {
      int r = e >> 6, c = e & 63;    // 8 consecutive hk of one row
      v8bf kv = *(const v8bf*)(kws + kvb + r * HEAD + c);
      *(v8bf*)&ks[r][c] = kv;
      v8bf vv = *(const v8bf*)(vws + kvb + r * HEAD + c);
      #pragma unroll
      for (int q = 0; q < 8; ++q) vsT[c + q][r] = vv[q];   // transpose store
    }
#endif
    if (j0 + 64 < TSEQ) {  // prefetch next K/V tiles (global_prefetch_b8)
      __builtin_prefetch(kws + kvb + 64 * HEAD + tid * 16, 0, 1);
      __builtin_prefetch(vws + kvb + 64 * HEAD + tid * 16, 0, 1);
    }
    __syncthreads();

    // S = Qs x Ks^T  (two 16x16 tiles per wave, 2 WMMAs each)
    #pragma unroll
    for (int it = 0; it < 2; ++it) {
      int ti = it ? ti1 : ti0;
      int tj = it ? tj1 : tj0;
      v8f sa = {};
      #pragma unroll
      for (int kc = 0; kc < 64; kc += 32) {
        v16bf a = ld_frag(&qs[ti * 16 + n][kc + k0h],
                          &qs[ti * 16 + n][kc + k0h + 16]);
        v16bf b = ld_frag(&ks[tj * 16 + n][kc + k0h],
                          &ks[tj * 16 + n][kc + k0h + 16]);
        sa = wmma_bf16(a, b, sa);
      }
      #pragma unroll
      for (int g = 0; g < 8; ++g)
        Ss[ti * 16 + g + khalf * 8][tj * 16 + n] = sa[g];
    }
    __syncthreads();

    // online softmax: one thread per row, float4 reads / v4bf writes
    if (tid < 64) {
      int r = tid;
      float rmax = mrow[r];
      #pragma unroll
      for (int c = 0; c < 64; c += 4) {
        float4 sv = *(const float4*)&Ss[r][c];
        rmax = fmaxf(rmax, fmaxf(fmaxf(sv.x, sv.y), fmaxf(sv.z, sv.w)));
      }
      float corr = __expf(mrow[r] - rmax);
      float s = 0.f;
      #pragma unroll
      for (int c = 0; c < 64; c += 4) {
        float4 sv = *(const float4*)&Ss[r][c];
        float p0 = __expf(sv.x - rmax), p1 = __expf(sv.y - rmax);
        float p2 = __expf(sv.z - rmax), p3 = __expf(sv.w - rmax);
        v4bf pk;
        pk[0] = (__bf16)p0; pk[1] = (__bf16)p1;
        pk[2] = (__bf16)p2; pk[3] = (__bf16)p3;
        *(v4bf*)&Ps[r][c] = pk;  // ds_store_b64
        s += (p0 + p1) + (p2 + p3);
      }
      lrow[r] = lrow[r] * corr + s;
      mrow[r] = rmax;
      crow[r] = corr;
    }
    __syncthreads();

    // rescale O and accumulate P x V
    #pragma unroll
    for (int it = 0; it < 2; ++it) {
      int ti = it ? ti1 : ti0;
      int tj = it ? tj1 : tj0;
      v8f o = accO[it];
      #pragma unroll
      for (int g = 0; g < 8; ++g)
        o[g] *= crow[ti * 16 + g + khalf * 8];
      #pragma unroll
      for (int kc = 0; kc < 64; kc += 32) {
        v16bf a = ld_frag(&Ps[ti * 16 + n][kc + k0h],
                          &Ps[ti * 16 + n][kc + k0h + 16]);
        v16bf b = ld_frag(&vsT[tj * 16 + n][kc + k0h],
                          &vsT[tj * 16 + n][kc + k0h + 16]);
        o = wmma_bf16(a, b, o);
      }
      accO[it] = o;
    }
  }
  __syncthreads();

  // normalize by 1/l and write out (f32, [B,T,H])
  #pragma unroll
  for (int it = 0; it < 2; ++it) {
    int ti = it ? ti1 : ti0;
    int tj = it ? tj1 : tj0;
    #pragma unroll
    for (int g = 0; g < 8; ++g) {
      int m = ti * 16 + g + khalf * 8;
      float invl = 1.f / lrow[m];
      out[qbase + (size_t)m * HEAD + tj * 16 + n] = accO[it][g] * invl;
    }
  }
}

// ---------------------------------------------------------------------------
extern "C" void kernel_launch(void* const* d_in, const int* in_sizes, int n_in,
                              void* d_out, int out_size, void* d_ws, size_t ws_size,
                              hipStream_t stream) {
  const float* x  = (const float*)d_in[0];
  const float* Wk = (const float*)d_in[1];
  const float* bk = (const float*)d_in[2];
  const float* Wq = (const float*)d_in[3];
  const float* bq = (const float*)d_in[4];
  const float* Wv = (const float*)d_in[5];
  float* out = (float*)d_out;

  const size_t nqkv = (size_t)BATCH * TSEQ * HEAD;
  __bf16* qws = (__bf16*)d_ws;
  __bf16* kws = qws + nqkv;
  __bf16* vws = kws + nqkv;

  dim3 blk(256);
  qkv_proj_kernel<<<(BATCH * TSEQ) / 64, blk, 0, stream>>>(
      x, Wq, bq, Wk, bk, Wv, qws, kws, vws);
  attn_kernel<<<BATCH * (TSEQ / 64), blk, 0, stream>>>(qws, kws, vws, out);
}
